// Transformer3DBlock_68066641707885
// MI455X (gfx1250) — matrix operand order repaired and checked by reading, not compiled, not run on hardware
//
#include <hip/hip_runtime.h>
#include <hip/hip_bf16.h>

typedef unsigned int  u32;
typedef unsigned short u16;
typedef __attribute__((ext_vector_type(16))) __bf16 v16bf;
typedef __attribute__((ext_vector_type(8)))  float  v8f;
typedef __attribute__((ext_vector_type(4)))  u32    v4u;

#define NPOS 32768   // D*H*W = 8*64*64
#define CDIM 128
#define BB   2

__device__ __forceinline__ u16 f2bf(float f) {
    u32 u = __float_as_uint(f);
    u32 r = (u + 0x7FFFu + ((u >> 16) & 1u)) >> 16;
    return (u16)r;
}
__device__ __forceinline__ float bf2f(u16 s) {
    return __uint_as_float(((u32)s) << 16);
}

union FragBF { v4u q[2]; v16bf v; };

// ---------------------------------------------------------------- utilities
__global__ void zero_scalars_kernel(float* p) {
    if (threadIdx.x < 2) p[threadIdx.x] = 0.f;
}

// FiLM: scale/shift[b,c] = t[b] . d{1,2}_w[c] + bias
__global__ void film_kernel(const float* __restrict__ t,
                            const float* __restrict__ d1w, const float* __restrict__ d1b,
                            const float* __restrict__ d2w, const float* __restrict__ d2b,
                            float* __restrict__ fsc, float* __restrict__ fsh) {
    int i = threadIdx.x;            // 256 threads: (b,c)
    int b = i >> 7, c = i & 127;
    const float* tb = t + b * 256;
    float a = 0.f, s = 0.f;
    for (int j = 0; j < 256; ++j) {
        float tv = tb[j];
        a += tv * d1w[c * 256 + j];
        s += tv * d2w[c * 256 + j];
    }
    fsc[i] = a + d1b[c];
    fsh[i] = s + d2b[c];
}

// pack conv weight f32 [M][IC][TAPS] -> bf16 [M][TAPS*IC]  (k = tap*IC+ic)
__global__ void pack_w_kernel(const float* __restrict__ src, u16* __restrict__ dst,
                              int M, int IC, int TAPS) {
    int i = blockIdx.x * 256 + threadIdx.x;
    if (i >= M * IC * TAPS) return;
    int m   = i / (IC * TAPS);
    int r   = i % (IC * TAPS);
    int ic  = r / TAPS;
    int tap = r % TAPS;
    dst[(size_t)m * (IC * TAPS) + tap * IC + ic] = f2bf(src[i]);
}

// LayerNorm over channel + affine gating: out = ((x-mu)*rstd*g[c]+be[c])*sc[b,c]+sh[b,c]
__global__ __launch_bounds__(256)
void ln_affine_kernel(const float* __restrict__ x,
                      const float* __restrict__ g, const float* __restrict__ be,
                      const float* __restrict__ sc, const float* __restrict__ sh,
                      u16* __restrict__ out) {
    int idx = blockIdx.x * 256 + threadIdx.x;       // B*NPOS threads
    int b = idx >> 15, pos = idx & (NPOS - 1);
    const float* xb = x + (size_t)b * CDIM * NPOS + pos;
    float s = 0.f, ss = 0.f;
    for (int c = 0; c < CDIM; ++c) {
        float v = xb[(size_t)c * NPOS];
        s += v; ss += v * v;
    }
    float mu = s * (1.f / CDIM);
    float var = ss * (1.f / CDIM) - mu * mu;
    float rstd = rsqrtf(var + 1e-5f);
    for (int c = 0; c < CDIM; ++c) {
        float v = (xb[(size_t)c * NPOS] - mu) * rstd;
        v = v * g[c] + be[c];
        float a = sc[b * CDIM + c];
        float h = sh ? sh[b * CDIM + c] : 0.f;
        out[((size_t)b * CDIM + c) * NPOS + pos] = f2bf(v * a + h);
    }
}

// ------------------------------------------------- WMMA implicit-GEMM conv
// M-tile 64 (two 16x16 tiles per wave sharing the B fragment), N-tile 64.
// Y[b][M][N] = sum_{tap,ic} Wp[m][tap*IC+ic] * X[b][ic][shift_tap(n)]  (+bias,relu,res)
template<int TAPS, bool RELU, bool HAS_BIAS, bool HAS_RES, bool WRITE_F32,
         bool WRITE_BF16, bool BSCALE>
__global__ __launch_bounds__(256)
void conv_gemm_kernel(const u16* __restrict__ X, const u16* __restrict__ Wp,
                      const float* __restrict__ bias, const float* __restrict__ res,
                      float* __restrict__ outF, u16* __restrict__ outB,
                      const float* __restrict__ bscale, int M, int IC) {
    __shared__ __align__(16) u16 lw[64 * 40];
    __shared__ __align__(16) u16 lx[64 * 40];
    int t = threadIdx.x, wave = t >> 5, lane = t & 31;
    int bz = blockIdx.z;
    int m0 = blockIdx.y * 64;
    int nt = blockIdx.x;                 // 512 tiles: (d, y)
    int d0 = nt >> 6, y0 = nt & 63;
    int n0 = nt * 64;
    int Ktot = TAPS * IC;
    const u16* Xb = X + (size_t)bz * IC * NPOS;
    int mi = wave >> 2, ni = wave & 3;

    v8f acc0 = {}, acc1 = {};
    int wm = t >> 2;                     // W-tile: 8 bf16 per thread (64x32)
    int wk = (t & 3) * 8;
    int xcol = t & 63;                   // X-tile load coords
    int kb4  = t >> 6;

    for (int tap = 0; tap < TAPS; ++tap) {
        int dz = 0, dy = 0, dx = 0;
        if (TAPS > 1) { dz = tap / 9 - 1; dy = (tap / 3) % 3 - 1; dx = tap % 3 - 1; }
        int dd = d0 + dz, yy = y0 + dy;
        bool rowOK = (dd >= 0) & (dd < 8) & (yy >= 0) & (yy < 64);
        int rowbase = (dd * 64 + yy) * 64;
        int xs = xcol + dx;
        bool ok = rowOK & (xs >= 0) & (xs < 64);
        for (int kc = 0; kc < IC; kc += 32) {
            __syncthreads();
            { // stage 64x32 W tile (16B per thread)
                const u16* src = Wp + (size_t)(m0 + wm) * Ktot + tap * IC + kc + wk;
                *(v4u*)&lw[wm * 40 + wk] = *(const v4u*)src;
            }
            { // stage 32k x 64n X tile, transposed with shift+mask
                #pragma unroll
                for (int it = 0; it < 8; ++it) {
                    int krow = it * 4 + kb4;
                    u16 v = 0;
                    if (ok) v = Xb[(size_t)(kc + krow) * NPOS + rowbase + xs];
                    lx[xcol * 40 + krow] = v;
                }
                if (ok && kc + 32 < IC)  // prefetch next k-step rows
                    __builtin_prefetch(&Xb[(size_t)(kc + 32 + kb4) * NPOS + rowbase + xs], 0, 0);
            }
            __syncthreads();
            FragBF A0, A1, Bf;
            { int m = mi * 16 + (lane & 15); int kb = (lane >> 4) * 8;
              A0.q[0] = *(const v4u*)&lw[m * 40 + kb];
              A0.q[1] = *(const v4u*)&lw[m * 40 + kb + 16];
              A1.q[0] = *(const v4u*)&lw[(m + 32) * 40 + kb];
              A1.q[1] = *(const v4u*)&lw[(m + 32) * 40 + kb + 16]; }
            { int n = ni * 16 + (lane & 15); int kb = (lane >> 4) * 8;
              Bf.q[0] = *(const v4u*)&lx[n * 40 + kb];
              Bf.q[1] = *(const v4u*)&lx[n * 40 + kb + 16]; }
            acc0 = __builtin_amdgcn_wmma_f32_16x16x32_bf16(
                       false, A0.v, false, Bf.v, (short)0, acc0, false, false);
            acc1 = __builtin_amdgcn_wmma_f32_16x16x32_bf16(
                       false, A1.v, false, Bf.v, (short)0, acc1, false, false);
        }
    }
    int ncol = n0 + ni * 16 + (lane & 15);
    #pragma unroll
    for (int s = 0; s < 2; ++s) {
        #pragma unroll
        for (int r = 0; r < 8; ++r) {
            int m = m0 + s * 32 + mi * 16 + r + 8 * (lane >> 4);
            float v = (s == 0) ? acc0[r] : acc1[r];
            if (HAS_BIAS) v += bias[m];
            if (RELU) v = fmaxf(v, 0.f);
            size_t oidx = ((size_t)bz * M + m) * NPOS + ncol;
            if (HAS_RES) v += res[oidx];
            if (WRITE_F32) outF[oidx] = v;
            if (WRITE_BF16) {
                float vb = BSCALE ? v * bscale[bz * M + m] : v;
                outB[oidx] = f2bf(vb);
            }
        }
    }
}

// depthwise 3x3x3 pad=1 over 384 channels, bf16 in/out, f32 weights [384][27]
__global__ __launch_bounds__(256)
void dw_kernel(const u16* __restrict__ in, const float* __restrict__ w,
               u16* __restrict__ out) {
    size_t i = (size_t)blockIdx.x * 256 + threadIdx.x;  // B*384*NPOS
    int pos = (int)(i & (NPOS - 1));
    int ch  = (int)((i >> 15) % 384);
    int b   = (int)((i >> 15) / 384);
    int d = pos >> 12, y = (pos >> 6) & 63, x = pos & 63;
    const u16* src = in + ((size_t)b * 384 + ch) * NPOS;
    const float* wc = w + ch * 27;
    float acc = 0.f;
    #pragma unroll
    for (int tap = 0; tap < 27; ++tap) {
        int dz = tap / 9 - 1, dy = (tap / 3) % 3 - 1, dx = tap % 3 - 1;
        int dd = d + dz, yy = y + dy, xx = x + dx;
        if (dd >= 0 && dd < 8 && yy >= 0 && yy < 64 && xx >= 0 && xx < 64)
            acc += wc[tap] * bf2f(src[(dd * 64 + yy) * 64 + xx]);
    }
    out[((size_t)b * 384 + ch) * NPOS + pos] = f2bf(acc);
}

// per-row L2 normalize (q,k = first 256 of 384 channels), in place
__global__ __launch_bounds__(256)
void l2norm_kernel(u16* __restrict__ qk) {
    __shared__ float red[256];
    int blk = blockIdx.x;                // B*256
    int b = blk >> 8, ch = blk & 255;
    u16* p = qk + ((size_t)b * 384 + ch) * NPOS;
    float ss = 0.f;
    for (int i = threadIdx.x; i < NPOS; i += 256) { float v = bf2f(p[i]); ss += v * v; }
    red[threadIdx.x] = ss; __syncthreads();
    for (int s = 128; s > 0; s >>= 1) {
        if (threadIdx.x < s) red[threadIdx.x] += red[threadIdx.x + s];
        __syncthreads();
    }
    float inv = 1.f / fmaxf(sqrtf(red[0]), 1e-12f);
    for (int i = threadIdx.x; i < NPOS; i += 256) p[i] = f2bf(bf2f(p[i]) * inv);
}

// channel attention: attn[b,h,n,m] = softmax_m( temp[h] * sum_f q[n,f]k[m,f] )
__global__ __launch_bounds__(256)
void attn_kernel(const u16* __restrict__ qk, const float* __restrict__ temp,
                 float* __restrict__ attn) {
    __shared__ float red[8 * 256];
    __shared__ float att[256];
    int t = threadIdx.x, wave = t >> 5, lane = t & 31;
    int bh = blockIdx.x;                // B*HEADS
    int b = bh >> 3, h = bh & 7;
    const u16* qrow = qk + ((size_t)b * 384 + h * 16 + (lane & 15)) * NPOS;
    const u16* krow = qk + ((size_t)b * 384 + 128 + h * 16 + (lane & 15)) * NPOS;
    v8f acc = {};
    int f0 = wave * 4096;
    for (int it = 0; it < 128; ++it) {
        int fb = f0 + it * 32;
        FragBF A, Bf;
        A.q[0]  = *(const v4u*)(qrow + fb + (lane >> 4) * 8);
        A.q[1]  = *(const v4u*)(qrow + fb + 16 + (lane >> 4) * 8);
        Bf.q[0] = *(const v4u*)(krow + fb + (lane >> 4) * 8);
        Bf.q[1] = *(const v4u*)(krow + fb + 16 + (lane >> 4) * 8);
        acc = __builtin_amdgcn_wmma_f32_16x16x32_bf16(
                  false, A.v, false, Bf.v, (short)0, acc, false, false);
    }
    #pragma unroll
    for (int r = 0; r < 8; ++r)
        red[wave * 256 + (r + 8 * (lane >> 4)) * 16 + (lane & 15)] = acc[r];
    __syncthreads();
    float s = 0.f;
    for (int w2 = 0; w2 < 8; ++w2) s += red[w2 * 256 + t];
    att[t] = s * temp[h];
    __syncthreads();
    if (t < 16) {
        float mx = -1e30f;
        for (int m = 0; m < 16; ++m) mx = fmaxf(mx, att[t * 16 + m]);
        float e[16]; float sum = 0.f;
        for (int m = 0; m < 16; ++m) { e[m] = __expf(att[t * 16 + m] - mx); sum += e[m]; }
        float inv = 1.f / sum;
        for (int m = 0; m < 16; ++m) attn[((size_t)bh * 16 + t) * 16 + m] = e[m] * inv;
    }
}

// out[b, h*16+n, f] = sum_m attn[b,h,n,m] * v[b, h*16+m, f]
__global__ __launch_bounds__(256)
void attn_out_kernel(const float* __restrict__ attn, const u16* __restrict__ qk,
                     u16* __restrict__ out) {
    int row = blockIdx.x;               // B*128
    int b = row >> 7, c = row & 127;
    int h = c >> 4, n = c & 15;
    int f = blockIdx.y * 256 + threadIdx.x;
    const float* a = attn + ((size_t)(b * 8 + h) * 16 + n) * 16;
    const u16* vb = qk + ((size_t)b * 384 + 256 + h * 16) * NPOS + f;
    float acc = 0.f;
    #pragma unroll
    for (int m = 0; m < 16; ++m) acc += a[m] * bf2f(vb[(size_t)m * NPOS]);
    out[(size_t)row * NPOS + f] = f2bf(acc);
}

// router MLP: softmax(256) -> FC512 -> LN -> gelu -> FC128 -> relu, share override
__global__ __launch_bounds__(512)
void router_kernel(const float* y1, const float* y2,
                   const float* w1a, const float* b1a, const float* lnwa, const float* lnba,
                   const float* w2a, const float* b2a,
                   const float* w1b, const float* b1b, const float* lnwb, const float* lnbb,
                   const float* w2b, const float* b2b,
                   float* s1, float* s2, float* scalars) {
    int r = blockIdx.x, b = blockIdx.y, t = threadIdx.x;
    const float* y   = (r == 0 ? y1 : y2) + b * 256;
    const float* w1  = r == 0 ? w1a : w1b;  const float* b1  = r == 0 ? b1a : b1b;
    const float* lnw = r == 0 ? lnwa : lnwb; const float* lnb = r == 0 ? lnba : lnbb;
    const float* w2  = r == 0 ? w2a : w2b;  const float* b2  = r == 0 ? b2a : b2b;
    float* sdst = (r == 0 ? s1 : s2) + b * 128;
    __shared__ float sm[256];
    __shared__ float z[512];
    __shared__ float red[512];
    red[t] = (t < 256) ? y[t] : -1e30f;
    __syncthreads();
    for (int s = 256; s > 0; s >>= 1) { if (t < s) red[t] = fmaxf(red[t], red[t + s]); __syncthreads(); }
    float mx = red[0]; __syncthreads();
    float e = (t < 256) ? __expf(y[t] - mx) : 0.f;
    red[t] = e; __syncthreads();
    for (int s = 256; s > 0; s >>= 1) { if (t < s) red[t] += red[t + s]; __syncthreads(); }
    float inv = 1.f / red[0];
    if (t < 256) sm[t] = e * inv;
    __syncthreads();
    { float a = b1[t]; const float* wr = w1 + (size_t)t * 256;
      for (int j = 0; j < 256; ++j) a += sm[j] * wr[j];
      z[t] = a; }
    __syncthreads();
    red[t] = z[t]; __syncthreads();
    for (int s = 256; s > 0; s >>= 1) { if (t < s) red[t] += red[t + s]; __syncthreads(); }
    float mu = red[0] * (1.f / 512.f); __syncthreads();
    float dv = z[t] - mu; red[t] = dv * dv; __syncthreads();
    for (int s = 256; s > 0; s >>= 1) { if (t < s) red[t] += red[t + s]; __syncthreads(); }
    float var = red[0] * (1.f / 512.f);
    float zn = (z[t] - mu) * rsqrtf(var + 1e-5f) * lnw[t] + lnb[t];
    float g = 0.5f * zn * (1.f + erff(zn * 0.70710678118f));
    __syncthreads();
    z[t] = g;
    __syncthreads();
    float sval = 0.f;
    if (t < 128) {
        float a = b2[t]; const float* wr = w2 + (size_t)t * 512;
        for (int j = 0; j < 512; ++j) a += z[j] * wr[j];
        sval = fmaxf(a, 0.f);
        sdst[t] = (t < 12) ? 1.0f : sval;   // SHARE = 12
    }
    red[t] = (t < 128) ? fabsf(sval) : 0.f;
    __syncthreads();
    for (int s = 256; s > 0; s >>= 1) { if (t < s) red[t] += red[t + s]; __syncthreads(); }
    if (t == 0) atomicAdd(&scalars[r], red[0] * (1.f / 256.f));
}

// ---------------------------------------------------------------- launch
extern "C" void kernel_launch(void* const* d_in, const int* in_sizes, int n_in,
                              void* d_out, int out_size, void* d_ws, size_t ws_size,
                              hipStream_t stream) {
    const float* x     = (const float*)d_in[0];
    const float* tvec  = (const float*)d_in[1];
    const float* y1    = (const float*)d_in[2];
    const float* y2    = (const float*)d_in[3];
    const float* ln1_w = (const float*)d_in[4];
    const float* ln1_b = (const float*)d_in[5];
    const float* d1_w  = (const float*)d_in[6];
    const float* d1_b  = (const float*)d_in[7];
    const float* d2_w  = (const float*)d_in[8];
    const float* d2_b  = (const float*)d_in[9];
    const float* temp  = (const float*)d_in[10];
    const float* qkv_w = (const float*)d_in[11];
    const float* dw_w  = (const float*)d_in[12];
    const float* proj_w= (const float*)d_in[13];
    const float* ln2_w = (const float*)d_in[14];
    const float* ln2_b = (const float*)d_in[15];
    const float* r1_w1 = (const float*)d_in[16];
    const float* r1_b1 = (const float*)d_in[17];
    const float* r1_lnw= (const float*)d_in[18];
    const float* r1_lnb= (const float*)d_in[19];
    const float* r1_w2 = (const float*)d_in[20];
    const float* r1_b2 = (const float*)d_in[21];
    const float* r2_w1 = (const float*)d_in[22];
    const float* r2_b1 = (const float*)d_in[23];
    const float* r2_lnw= (const float*)d_in[24];
    const float* r2_lnb= (const float*)d_in[25];
    const float* r2_w2 = (const float*)d_in[26];
    const float* r2_b2 = (const float*)d_in[27];
    const float* moe1_w= (const float*)d_in[28];
    const float* moe1_b= (const float*)d_in[29];
    const float* moe2_w= (const float*)d_in[30];
    const float* moe2_b= (const float*)d_in[31];

    char* ws = (char*)d_ws;
    const size_t OFF_A   = 0;                       // 50.3MB: qkv bf16 -> x_res f32
    const size_t OFF_A2  = 50331648;                // 50.3MB: qkv_dw bf16 -> h1 f32
    const size_t OFF_C1  = 100663296;               // 16.8MB: xm bf16 -> xg bf16
    const size_t OFF_C2  = 117440512;               // 16.8MB: attn_out bf16 -> h1g bf16
    const size_t OFF_WQ  = 134217728;               // packed weights & small
    const size_t OFF_WP  = OFF_WQ + 98304;
    const size_t OFF_WM1 = OFF_WP + 32768;
    const size_t OFF_WM2 = OFF_WM1 + 884736;
    const size_t OFF_ATT = OFF_WM2 + 884736;
    const size_t OFF_FSC = OFF_ATT + 16384;
    const size_t OFF_FSH = OFF_FSC + 1024;
    const size_t OFF_S1  = OFF_FSH + 1024;
    const size_t OFF_S2  = OFF_S1 + 1024;

    u16*   qkv   = (u16*)(ws + OFF_A);
    u16*   qkvdw = (u16*)(ws + OFF_A2);
    u16*   xm    = (u16*)(ws + OFF_C1);
    u16*   aout  = (u16*)(ws + OFF_C2);
    float* xres  = (float*)(ws + OFF_A);
    float* h1    = (float*)(ws + OFF_A2);
    u16*   xg    = (u16*)(ws + OFF_C1);
    u16*   h1g   = (u16*)(ws + OFF_C2);
    u16*   wq    = (u16*)(ws + OFF_WQ);
    u16*   wp    = (u16*)(ws + OFF_WP);
    u16*   wm1   = (u16*)(ws + OFF_WM1);
    u16*   wm2   = (u16*)(ws + OFF_WM2);
    float* attn  = (float*)(ws + OFF_ATT);
    float* fsc   = (float*)(ws + OFF_FSC);
    float* fsh   = (float*)(ws + OFF_FSH);
    float* s1    = (float*)(ws + OFF_S1);
    float* s2    = (float*)(ws + OFF_S2);
    float* outF  = (float*)d_out;
    float* scal  = outF + (size_t)BB * CDIM * NPOS;  // 8388608

    zero_scalars_kernel<<<1, 32, 0, stream>>>(scal);
    film_kernel<<<1, 256, 0, stream>>>(tvec, d1_w, d1_b, d2_w, d2_b, fsc, fsh);
    pack_w_kernel<<<192, 256, 0, stream>>>(qkv_w, wq, 384, 128, 1);
    pack_w_kernel<<<64, 256, 0, stream>>>(proj_w, wp, 128, 128, 1);
    pack_w_kernel<<<1728, 256, 0, stream>>>(moe1_w, wm1, 128, 128, 27);
    pack_w_kernel<<<1728, 256, 0, stream>>>(moe2_w, wm2, 128, 128, 27);

    // attention branch
    ln_affine_kernel<<<256, 256, 0, stream>>>(x, ln1_w, ln1_b, fsc, fsh, xm);
    conv_gemm_kernel<1, false, false, false, false, true, false>
        <<<dim3(512, 6, BB), 256, 0, stream>>>(xm, wq, nullptr, nullptr,
                                               nullptr, qkv, nullptr, 384, 128);
    dw_kernel<<<98304, 256, 0, stream>>>(qkv, dw_w, qkvdw);
    l2norm_kernel<<<BB * 256, 256, 0, stream>>>(qkvdw);
    attn_kernel<<<BB * 8, 256, 0, stream>>>(qkvdw, temp, attn);
    attn_out_kernel<<<dim3(BB * CDIM, NPOS / 256), 256, 0, stream>>>(attn, qkvdw, aout);
    conv_gemm_kernel<1, false, false, true, true, false, false>
        <<<dim3(512, 2, BB), 256, 0, stream>>>(aout, wp, nullptr, x,
                                               xres, nullptr, nullptr, 128, 128);

    // MoE branch
    router_kernel<<<dim3(2, BB), 512, 0, stream>>>(
        y1, y2, r1_w1, r1_b1, r1_lnw, r1_lnb, r1_w2, r1_b2,
        r2_w1, r2_b1, r2_lnw, r2_lnb, r2_w2, r2_b2, s1, s2, scal);
    ln_affine_kernel<<<256, 256, 0, stream>>>(xres, ln2_w, ln2_b, s1, nullptr, xg);
    conv_gemm_kernel<27, true, true, false, true, true, true>
        <<<dim3(512, 2, BB), 256, 0, stream>>>(xg, wm1, moe1_b, nullptr,
                                               h1, h1g, s2, 128, 128);
    conv_gemm_kernel<27, false, true, true, true, false, false>
        <<<dim3(512, 2, BB), 256, 0, stream>>>(h1g, wm2, moe2_b, h1,
                                               outF, nullptr, nullptr, 128, 128);
    (void)in_sizes; (void)n_in; (void)out_size; (void)ws_size;
}
